// MultiHeadAttentionSpatial_1821066133788
// MI455X (gfx1250) — matrix operand order repaired and checked by reading, not compiled, not run on hardware
//
#include <hip/hip_runtime.h>
#include <hip/hip_bf16.h>

// Problem constants (from reference): B=32, L=512, D=768, H=12, SD=5, DH=64
#define B_  32
#define L_  512
#define D_  768
#define H_  12
#define SD_ 5
#define DH_ 64
#define CW_ (H_ * (SD_ + 1))   // 72

typedef __bf16 bf16_t;
typedef __attribute__((ext_vector_type(16))) __bf16 v16bf;
typedef __attribute__((ext_vector_type(8)))  float  v8f;

union Frag128x2 { v16bf v; uint4 q[2]; };

// ---------------------------------------------------------------------------
// CDNA5 async copy: global -> LDS, 16 bytes per lane, tracked by ASYNCcnt.
// (CDNA5 ISA; inline asm is the toolchain-portable path.)
__device__ inline void async_copy16(unsigned lds_off, const void* gptr) {
  asm volatile("global_load_async_to_lds_b128 %0, %1, off"
               :: "v"(lds_off), "v"((unsigned long long)(size_t)gptr)
               : "memory");
}
__device__ inline void async_wait0() {
  asm volatile("s_wait_asynccnt 0x0" ::: "memory");
}
__device__ inline unsigned lds_addr32(const void* p) {
  // Flat LDS-aperture addresses carry the wave-relative LDS offset in [31:0].
  return (unsigned)(size_t)p;
}

// ---------------------------------------------------------------------------
// WMMA fragment loaders (wave32, 16-bit operands, BK = 32)
// A 16x32 bf16 (ISA 7.12.2): lanes0-15 M=0..15 hold K {0..7,16..23},
// lanes16-31 hold K {8..15,24..31}.  LDS tile row-major [m][32].
__device__ inline v16bf load_a_frag(const bf16_t* tile, int row0, int lane) {
  const int m     = row0 + (lane & 15);
  const int khalf = (lane >> 4) * 8;          // 0 or 8
  const char* base = (const char*)(tile + m * 32);
  Frag128x2 u;
  u.q[0] = *(const uint4*)(base + khalf * 2);
  u.q[1] = *(const uint4*)(base + (khalf + 16) * 2);
  return u.v;
}
// B 32x16 bf16: lane = column N; lanes0-15 K {0..7,16..23}, lanes16-31 K {8..15,24..31}.
// LDS tile stored transposed [n][32] so K runs contiguous.
__device__ inline v16bf load_b_frag(const bf16_t* tileT, int col0, int lane) {
  const int n     = col0 + (lane & 15);
  const int kbase = (lane >> 4) * 8;          // 0 or 8
  const char* base = (const char*)(tileT + n * 32);
  Frag128x2 u;
  u.q[0] = *(const uint4*)(base + kbase * 2);
  u.q[1] = *(const uint4*)(base + (kbase + 16) * 2);
  return u.v;
}
__device__ inline v8f wmma_bf16(v16bf a, v16bf b, v8f c) {
  return __builtin_amdgcn_wmma_f32_16x16x32_bf16(false, a, false, b,
                                                 (short)0, c, false, false);
}

// ---------------------------------------------------------------------------
// GEMM: Y[M,N] = X[M,K]*W[K,N] + bias.  64x64 tile, BK=32, 4 waves.
// BF16_OUT selects bf16 (qh/kh/vh) vs f32 (final projection) output.
template <bool BF16_OUT>
__global__ __launch_bounds__(128) void gemm_bias(
    const float* __restrict__ X, const float* __restrict__ W,
    const float* __restrict__ bias, void* __restrict__ Yv,
    int M, int N, int K) {
  __shared__ __align__(16) bf16_t As[64 * 32];
  __shared__ __align__(16) bf16_t Bt[64 * 32];   // [n][k]
  const int tid = threadIdx.x, lane = tid & 31, wave = tid >> 5;
  const int n0 = blockIdx.x * 64, m0 = blockIdx.y * 64;

  v8f acc[4];
#pragma unroll
  for (int i = 0; i < 4; ++i)
#pragma unroll
    for (int r = 0; r < 8; ++r) acc[i][r] = 0.f;

  for (int kk = 0; kk < K; kk += 32) {
#pragma unroll
    for (int it = 0; it < 4; ++it) {             // A: 64x32 f32 -> bf16
      int i = it * 128 + tid;
      int r = i >> 3, c = (i & 7) * 4;
      float4 f = *(const float4*)(X + (size_t)(m0 + r) * K + kk + c);
      bf16_t* d = As + r * 32 + c;
      d[0] = (bf16_t)f.x; d[1] = (bf16_t)f.y; d[2] = (bf16_t)f.z; d[3] = (bf16_t)f.w;
    }
#pragma unroll
    for (int it = 0; it < 4; ++it) {             // B: W[k][n] -> Bt[n][k]
      int i = it * 128 + tid;
      int k = i >> 4, c = (i & 15) * 4;
      float4 f = *(const float4*)(W + (size_t)(kk + k) * N + n0 + c);
      Bt[(c + 0) * 32 + k] = (bf16_t)f.x;
      Bt[(c + 1) * 32 + k] = (bf16_t)f.y;
      Bt[(c + 2) * 32 + k] = (bf16_t)f.z;
      Bt[(c + 3) * 32 + k] = (bf16_t)f.w;
    }
    __syncthreads();
    v16bf a = load_a_frag(As, wave * 16, lane);
#pragma unroll
    for (int nt = 0; nt < 4; ++nt) {
      v16bf b = load_b_frag(Bt, nt * 16, lane);
      acc[nt] = wmma_bf16(a, b, acc[nt]);
    }
    __syncthreads();
  }
  const int rbase = m0 + wave * 16 + ((lane >> 4) << 3);
  const int cl = lane & 15;
#pragma unroll
  for (int nt = 0; nt < 4; ++nt) {
    const int col = n0 + nt * 16 + cl;
    const float bv = bias[col];
#pragma unroll
    for (int r = 0; r < 8; ++r) {
      float o = acc[nt][r] + bv;
      if (BF16_OUT) ((bf16_t*)Yv)[(size_t)(rbase + r) * N + col] = (bf16_t)o;
      else          ((float*)Yv)[(size_t)(rbase + r) * N + col] = o;
    }
  }
}

// ---------------------------------------------------------------------------
// Kernel 2: sw[b,l,j] = sum_d (q[b,l,d]+txt[b,d]) * Wcond[d,j] + bcond[j]
__global__ __launch_bounds__(128) void cond_sw_kernel(
    const float* __restrict__ q, const float* __restrict__ txt,
    const float* __restrict__ Wc, const float* __restrict__ bc,
    float* __restrict__ swout) {
  const int row = blockIdx.x;          // b*L + l
  const int b = row / L_;
  __shared__ float xs[D_];
#pragma unroll
  for (int it = 0; it < 6; ++it) {
    int i = it * 128 + threadIdx.x;
    xs[i] = q[(size_t)row * D_ + i] + txt[(size_t)b * D_ + i];
  }
  __syncthreads();
  if (threadIdx.x < CW_) {
    const int j = threadIdx.x;
    float acc = bc[j];
    for (int d = 0; d < D_; ++d) acc = fmaf(xs[d], Wc[(size_t)d * CW_ + j], acc);
    swout[(size_t)row * CW_ + j] = acc;
  }
}

// ---------------------------------------------------------------------------
// Kernel 3: S[b,h,l,t] = 0.125 * sum_k qh[b,l,h,k]*kh[b,t,h,k]
// qh/kh are bf16 -> both tiles staged with GLOBAL_LOAD_ASYNC_TO_LDS_B128
// (pure byte copy, zero VALU conversion; ASYNCcnt + barrier for visibility).
__global__ __launch_bounds__(128) void qk_scores(
    const bf16_t* __restrict__ qh, const bf16_t* __restrict__ kh,
    float* __restrict__ fused) {
  __shared__ __align__(16) bf16_t As[64 * 32];
  __shared__ __align__(16) bf16_t Bt[64 * 32];   // Bt[t][k] = kh[t][k]
  const int tid = threadIdx.x, lane = tid & 31, wave = tid >> 5;
  const int t0 = blockIdx.x * 64, l0 = blockIdx.y * 64;
  const int bh = blockIdx.z, b = bh / H_, h = bh % H_;
  const bf16_t* qbase = qh + (size_t)b * L_ * D_ + h * DH_;
  const bf16_t* kbase = kh + (size_t)b * L_ * D_ + h * DH_;
  const unsigned asA = lds_addr32(As);
  const unsigned asB = lds_addr32(Bt);

  v8f acc[4];
#pragma unroll
  for (int i = 0; i < 4; ++i)
#pragma unroll
    for (int r = 0; r < 8; ++r) acc[i][r] = 0.f;

  for (int kk = 0; kk < DH_; kk += 32) {
    // 64 rows x 64 bytes per tile; 4 lanes/row x 16B; 2 issues per tile.
#pragma unroll
    for (int it = 0; it < 2; ++it) {
      int id = it * 128 + tid;
      int r = id >> 2, ch = (id & 3) * 16;
      async_copy16(asA + r * 64 + ch,
                   (const char*)(qbase + (size_t)(l0 + r) * D_ + kk) + ch);
      async_copy16(asB + r * 64 + ch,
                   (const char*)(kbase + (size_t)(t0 + r) * D_ + kk) + ch);
    }
    async_wait0();
    __syncthreads();
    v16bf a = load_a_frag(As, wave * 16, lane);
#pragma unroll
    for (int nt = 0; nt < 4; ++nt) {
      v16bf bfr = load_b_frag(Bt, nt * 16, lane);
      acc[nt] = wmma_bf16(a, bfr, acc[nt]);
    }
    __syncthreads();
  }
  float* out = fused + ((size_t)bh * L_ + l0) * L_ + t0;
  const int rbase = wave * 16 + ((lane >> 4) << 3);
  const int cl = lane & 15;
#pragma unroll
  for (int nt = 0; nt < 4; ++nt)
#pragma unroll
    for (int r = 0; r < 8; ++r)
      out[(size_t)(rbase + r) * L_ + nt * 16 + cl] = acc[nt][r] * 0.125f;
}

// ---------------------------------------------------------------------------
// Kernel 4: per (b,l), all 12 heads: sigmoid gate + log + mask + softmax,
// in place on fused_attn.  pairwise_locs row staged once (12x reuse).
__global__ __launch_bounds__(256) void fuse_softmax_kernel(
    float* __restrict__ fused, const float* __restrict__ ploc,
    const float* __restrict__ sw, const int* __restrict__ mask) {
  const int row = blockIdx.x;          // b*L + l
  const int b = row / L_, l = row % L_;
  const int tid = threadIdx.x;
  __shared__ float pl[L_ * SD_];       // 10 KB
  __shared__ float swr[CW_];
  __shared__ float buf[L_];
  __shared__ float red[256];
  __shared__ unsigned char mk[L_];

#pragma unroll
  for (int it = 0; it < 10; ++it)
    pl[it * 256 + tid] = ploc[(size_t)row * L_ * SD_ + it * 256 + tid];
  if (tid < CW_) swr[tid] = sw[(size_t)row * CW_ + tid];
#pragma unroll
  for (int it = 0; it < 2; ++it) {
    int i = it * 256 + tid;
    mk[i] = (mask[(size_t)b * L_ + i] != 0);
  }
  __syncthreads();

  for (int h = 0; h < H_; ++h) {
    float* Srow = fused + (((size_t)b * H_ + h) * L_ + l) * L_;
    const float bias = swr[h * (SD_ + 1)];
    const float w0 = swr[h * (SD_ + 1) + 1], w1 = swr[h * (SD_ + 1) + 2];
    const float w2 = swr[h * (SD_ + 1) + 3], w3 = swr[h * (SD_ + 1) + 4];
    const float w4 = swr[h * (SD_ + 1) + 5];

    float lmax = -3.4e38f;
#pragma unroll
    for (int it = 0; it < 2; ++it) {
      int t = it * 256 + tid;
      float logit;
      if (mk[t]) {
        logit = -__builtin_inff();
      } else {
        const float* p = pl + t * SD_;
        float z = fmaf(w0, p[0], fmaf(w1, p[1], fmaf(w2, p[2],
                  fmaf(w3, p[3], fmaf(w4, p[4], bias)))));
        float sg = 1.f / (1.f + __expf(-z));
        logit = __logf(fmaxf(sg, 1e-6f)) + Srow[t];
      }
      buf[t] = logit;
      lmax = fmaxf(lmax, logit);
    }
    red[tid] = lmax; __syncthreads();
    for (int s = 128; s > 0; s >>= 1) {
      if (tid < s) red[tid] = fmaxf(red[tid], red[tid + s]);
      __syncthreads();
    }
    const float m = red[0]; __syncthreads();

    float lsum = 0.f;
#pragma unroll
    for (int it = 0; it < 2; ++it) {
      int t = it * 256 + tid;
      float e = __expf(buf[t] - m); buf[t] = e; lsum += e;
    }
    red[tid] = lsum; __syncthreads();
    for (int s = 128; s > 0; s >>= 1) {
      if (tid < s) red[tid] += red[tid + s];
      __syncthreads();
    }
    const float inv = 1.f / red[0]; __syncthreads();
#pragma unroll
    for (int it = 0; it < 2; ++it) {
      int t = it * 256 + tid;
      Srow[t] = buf[t] * inv;
    }
    __syncthreads();
  }
}

// ---------------------------------------------------------------------------
// Kernel 5: out[b,l,h,v] = sum_t fused[b,h,l,t] * vh[b,t,h,v]
// A = f32 probabilities (VALU cvt); B = bf16 vh, direct transpose staging.
__global__ __launch_bounds__(128) void av_gemm(
    const float* __restrict__ fused, const bf16_t* __restrict__ vh,
    float* __restrict__ aout) {
  __shared__ __align__(16) bf16_t As[64 * 32];
  __shared__ __align__(16) bf16_t Bt[64 * 32];   // Bt[n][k] = vh[b,kk+k,h,n]
  const int tid = threadIdx.x, lane = tid & 31, wave = tid >> 5;
  const int l0 = blockIdx.x * 64;
  const int bh = blockIdx.y, b = bh / H_, h = bh % H_;
  const float* Abase = fused + ((size_t)bh * L_ + l0) * L_;

  v8f acc[4];
#pragma unroll
  for (int i = 0; i < 4; ++i)
#pragma unroll
    for (int r = 0; r < 8; ++r) acc[i][r] = 0.f;

  for (int kk = 0; kk < L_; kk += 32) {
#pragma unroll
    for (int it = 0; it < 4; ++it) {
      int i = it * 128 + tid;
      int r = i >> 3, c = (i & 7) * 4;
      float4 f = *(const float4*)(Abase + (size_t)r * L_ + kk + c);
      bf16_t* d = As + r * 32 + c;
      d[0] = (bf16_t)f.x; d[1] = (bf16_t)f.y; d[2] = (bf16_t)f.z; d[3] = (bf16_t)f.w;
    }
#pragma unroll
    for (int it = 0; it < 2; ++it) {             // 32 rows x 8 chunks of 8 bf16
      int id = it * 128 + tid;
      int k = id >> 3, c = (id & 7) * 8;
      const bf16_t* src = vh + ((size_t)b * L_ + kk + k) * D_ + h * DH_ + c;
#pragma unroll
      for (int j = 0; j < 8; ++j) Bt[(c + j) * 32 + k] = src[j];
    }
    __syncthreads();
    v16bf a = load_a_frag(As, wave * 16, lane);
#pragma unroll
    for (int nt = 0; nt < 4; ++nt) {
      v16bf bfr = load_b_frag(Bt, nt * 16, lane);
      acc[nt] = wmma_bf16(a, bfr, acc[nt]);
    }
    __syncthreads();
  }
  const int rbase = wave * 16 + ((lane >> 4) << 3);
  const int cl = lane & 15;
#pragma unroll
  for (int nt = 0; nt < 4; ++nt)
#pragma unroll
    for (int r = 0; r < 8; ++r)
      aout[((size_t)b * L_ + l0 + rbase + r) * D_ + h * DH_ + nt * 16 + cl] = acc[nt][r];
}

// ---------------------------------------------------------------------------
// Kernel 7: residual add + LayerNorm -> y
__global__ __launch_bounds__(256) void ln_kernel(
    const float* __restrict__ proj, const float* __restrict__ resid,
    const float* __restrict__ gamma, const float* __restrict__ beta,
    float* __restrict__ y) {
  const int row = blockIdx.x;
  const int tid = threadIdx.x;
  __shared__ float xb[D_];
  __shared__ float red[256];
  float s = 0.f;
#pragma unroll
  for (int it = 0; it < 3; ++it) {
    int i = it * 256 + tid;
    float x = proj[(size_t)row * D_ + i] + resid[(size_t)row * D_ + i];
    xb[i] = x; s += x;
  }
  red[tid] = s; __syncthreads();
  for (int st = 128; st > 0; st >>= 1) {
    if (tid < st) red[tid] += red[tid + st];
    __syncthreads();
  }
  const float mu = red[0] / D_; __syncthreads();
  float v = 0.f;
#pragma unroll
  for (int it = 0; it < 3; ++it) {
    int i = it * 256 + tid;
    float d = xb[i] - mu; v += d * d;
  }
  red[tid] = v; __syncthreads();
  for (int st = 128; st > 0; st >>= 1) {
    if (tid < st) red[tid] += red[tid + st];
    __syncthreads();
  }
  const float rstd = rsqrtf(red[0] / D_ + 1e-5f);
#pragma unroll
  for (int it = 0; it < 3; ++it) {
    int i = it * 256 + tid;
    y[(size_t)row * D_ + i] = (xb[i] - mu) * rstd * gamma[i] + beta[i];
  }
}

// ---------------------------------------------------------------------------
extern "C" void kernel_launch(void* const* d_in, const int* in_sizes, int n_in,
                              void* d_out, int out_size, void* d_ws, size_t ws_size,
                              hipStream_t stream) {
  (void)in_sizes; (void)n_in; (void)out_size; (void)ws_size;
  const float* q     = (const float*)d_in[0];
  const float* k     = (const float*)d_in[1];
  const float* v     = (const float*)d_in[2];
  const float* ploc  = (const float*)d_in[3];
  const float* txt   = (const float*)d_in[4];
  const int*   mask  = (const int*)d_in[5];
  const float* Wq    = (const float*)d_in[6];
  const float* bq    = (const float*)d_in[7];
  const float* Wk    = (const float*)d_in[8];
  const float* bk    = (const float*)d_in[9];
  const float* Wv    = (const float*)d_in[10];
  const float* bv    = (const float*)d_in[11];
  const float* Wfc   = (const float*)d_in[12];
  const float* bfc   = (const float*)d_in[13];
  const float* Wcond = (const float*)d_in[14];
  const float* bcond = (const float*)d_in[15];
  const float* gamma = (const float*)d_in[16];
  const float* beta  = (const float*)d_in[17];

  float* y     = (float*)d_out;                       // (B,L,D)
  float* fused = y + (size_t)B_ * L_ * D_;            // (B,H,L,L)

  const size_t BLD = (size_t)B_ * L_ * D_;
  char* ws = (char*)d_ws;
  bf16_t* qh  = (bf16_t*)ws;                          // BLD bf16
  bf16_t* kh  = qh + BLD;
  bf16_t* vh  = kh + BLD;
  float* aout = (float*)(ws + 3 * BLD * sizeof(bf16_t));
  float* swb  = (float*)(ws + 3 * BLD * sizeof(bf16_t) + BLD * sizeof(float));
  float* proj = (float*)ws;   // reuse qh/kh region (free after attention)

  const dim3 blkG(128), grdG(D_ / 64, (B_ * L_) / 64);
  gemm_bias<true><<<grdG, blkG, 0, stream>>>(q, Wq, bq, qh, B_ * L_, D_, D_);
  gemm_bias<true><<<grdG, blkG, 0, stream>>>(k, Wk, bk, kh, B_ * L_, D_, D_);
  gemm_bias<true><<<grdG, blkG, 0, stream>>>(v, Wv, bv, vh, B_ * L_, D_, D_);
  cond_sw_kernel<<<dim3(B_ * L_), dim3(128), 0, stream>>>(q, txt, Wcond, bcond, swb);
  qk_scores<<<dim3(L_ / 64, L_ / 64, B_ * H_), dim3(128), 0, stream>>>(qh, kh, fused);
  fuse_softmax_kernel<<<dim3(B_ * L_), dim3(256), 0, stream>>>(fused, ploc, swb, mask);
  av_gemm<<<dim3(L_ / 64, B_ * H_), dim3(128), 0, stream>>>(fused, vh, aout);
  gemm_bias<false><<<grdG, blkG, 0, stream>>>(aout, Wfc, bfc, proj, B_ * L_, D_, D_);
  ln_kernel<<<dim3(B_ * L_), dim3(256), 0, stream>>>(proj, q, gamma, beta, y);
}
